// LSMDecoder_11836929868281
// MI455X (gfx1250) — matrix operand order repaired and checked by reading, not compiled, not run on hardware
//
#include <hip/hip_runtime.h>
#include <hip/hip_bf16.h>

typedef __attribute__((ext_vector_type(16))) _Float16 v16h;
typedef __attribute__((ext_vector_type(8)))  float    v8f;

#define N_NODES 8192
#define D_IN    512
#define D_LAT   128
#define KDIM    512          // 4 * D_LAT packed contraction axis
#define SMALL_EPS 1e-16f

// ---------------------------------------------------------------------------
// Kernel 1: column softmax of the three [512 x 128] weight matrices.
// One block per column (384 blocks, 256 threads; each thread owns 2 rows).
// ---------------------------------------------------------------------------
__global__ void lsm_softmax_cols(const float* __restrict__ Wz,
                                 const float* __restrict__ Wg,
                                 const float* __restrict__ Wd,
                                 float* __restrict__ T) {
    const int m = blockIdx.x >> 7;          // which matrix (0..2)
    const int c = blockIdx.x & 127;         // column (0..127)
    const float* W = (m == 0) ? Wz : (m == 1) ? Wg : Wd;
    const int t = threadIdx.x;              // 256 threads
    float v0 = W[(size_t)t         * D_LAT + c];
    float v1 = W[(size_t)(t + 256) * D_LAT + c];

    __shared__ float red[256];
    red[t] = fmaxf(v0, v1);
    __syncthreads();
    for (int s = 128; s > 0; s >>= 1) {
        if (t < s) red[t] = fmaxf(red[t], red[t + s]);
        __syncthreads();
    }
    const float mx = red[0];
    __syncthreads();

    float e0 = __expf(v0 - mx);
    float e1 = __expf(v1 - mx);
    red[t] = e0 + e1;
    __syncthreads();
    for (int s = 128; s > 0; s >>= 1) {
        if (t < s) red[t] += red[t + s];
        __syncthreads();
    }
    const float inv = 1.0f / red[0];
    float* Tm = T + (size_t)m * (D_IN * D_LAT);
    Tm[(size_t)t         * D_LAT + c] = e0 * inv;
    Tm[(size_t)(t + 256) * D_LAT + c] = e1 * inv;
}

// ---------------------------------------------------------------------------
// Kernel 2: projections zd = z@Tz, gd_raw = gamma@Tg, dd_raw = delta@Td.
// Block = 128 threads, handles 16 rows x 128 cols of one matrix; the 16x512
// input slab is staged in LDS (32 KB), T columns stream coalesced from L2.
// ---------------------------------------------------------------------------
__global__ void lsm_proj(const float* __restrict__ z,
                         const float* __restrict__ gamma,
                         const float* __restrict__ delta,
                         const float* __restrict__ T,
                         float* __restrict__ zd,
                         float* __restrict__ gd,
                         float* __restrict__ dd) {
    __shared__ float xs[16 * D_IN];         // 32 KB
    const int m  = blockIdx.y;
    const int i0 = blockIdx.x * 16;
    const float* X = (m == 0) ? z  : (m == 1) ? gamma : delta;
    float*       O = (m == 0) ? zd : (m == 1) ? gd    : dd;
    const int t = threadIdx.x;              // 128 threads

    for (int idx = t; idx < 16 * D_IN; idx += 128)
        xs[idx] = X[(size_t)i0 * D_IN + idx];
    __syncthreads();

    float acc[16];
#pragma unroll
    for (int r = 0; r < 16; ++r) acc[r] = 0.0f;

    const float* Tm = T + (size_t)m * (D_IN * D_LAT);
    for (int k = 0; k < D_IN; ++k) {
        const float tv = Tm[(size_t)k * D_LAT + t];
#pragma unroll
        for (int r = 0; r < 16; ++r)
            acc[r] = fmaf(xs[r * D_IN + k], tv, acc[r]);
    }
#pragma unroll
    for (int r = 0; r < 16; ++r)
        O[(size_t)(i0 + r) * D_LAT + t] = acc[r];
}

// ---------------------------------------------------------------------------
// Kernel 3: per-node scalar terms
//   r_i = bias - w_gam*N * sum_k zd^2/gd ,  c_j = -w_del*N * sum_k zd^2/dd
// ---------------------------------------------------------------------------
__global__ void lsm_rowcol(const float* __restrict__ zd,
                           const float* __restrict__ gd_raw,
                           const float* __restrict__ dd_raw,
                           const float* __restrict__ bias_p,
                           const float* __restrict__ wgam_p,
                           const float* __restrict__ wdel_p,
                           float* __restrict__ rvec,
                           float* __restrict__ cvec) {
    const int i = blockIdx.x;
    const int t = threadIdx.x;              // 128 threads
    const float zv = zd[(size_t)i * D_LAT + t];
    const float g  = gd_raw[(size_t)i * D_LAT + t] + SMALL_EPS;
    const float d  = dd_raw[(size_t)i * D_LAT + t] + SMALL_EPS;
    const float z2 = zv * zv;

    __shared__ float sg[128];
    __shared__ float sd[128];
    sg[t] = z2 / g;
    sd[t] = z2 / d;
    __syncthreads();
    for (int s = 64; s > 0; s >>= 1) {
        if (t < s) { sg[t] += sg[t + s]; sd[t] += sd[t + s]; }
        __syncthreads();
    }
    if (t == 0) {
        const float n = (float)N_NODES;
        rvec[i] = bias_p[0] - wgam_p[0] * n * sg[0];
        cvec[i] = -wdel_p[0] * n * sd[0];
    }
}

// ---------------------------------------------------------------------------
// Kernel 4: build WMMA-swizzled f16 operand matrices A, B  [8192 x 512].
// Per 16-row panel p and K-block kb (32 wide), lane L loads 16 contiguous
// halves at offset p*8192 + kb*512 + L*16.  A and B both use the contiguous
// lane layout (lanes 0-15: K 0-15; lanes 16-31: K 16-31).
// sqrt-split scaling keeps both operands well inside f16 range.
// ---------------------------------------------------------------------------
__global__ void lsm_pack(const float* __restrict__ zd,
                         const float* __restrict__ gd_raw,
                         const float* __restrict__ dd_raw,
                         const float* __restrict__ wgam_p,
                         const float* __restrict__ wdel_p,
                         _Float16* __restrict__ Apk,
                         _Float16* __restrict__ Bpk) {
    const size_t per = (size_t)N_NODES * KDIM;
    size_t tid = (size_t)blockIdx.x * blockDim.x + threadIdx.x;
    const bool isB = tid >= per;
    const size_t idx = isB ? tid - per : tid;

    const int p  = (int)(idx >> 13);        // 16-row panel (8192 halves each)
    const int q  = (int)(idx & 8191);
    const int kb = q >> 9;                  // K-block of 32 (512 halves each)
    const int w  = q & 511;
    const int L  = w >> 4;                  // lane
    const int e  = w & 15;                  // element within lane's v16h

    int r, klocal;
    if (!isB) {                             // A-operand swizzle
        if (L < 16) { r = L;      klocal = e;      }
        else        { r = L - 16; klocal = e + 16; }
    } else {                                // B-operand swizzle
        if (L < 16) { r = L;      klocal = e;      }
        else        { r = L - 16; klocal = e + 16; }
    }
    const int kk = kb * 32 + klocal;        // 0..511
    const int i  = p * 16 + r;              // node index (row of A / col of B)
    const int s  = kk >> 7;                 // segment 0..3
    const int k  = kk & 127;                // latent index

    const float n  = (float)N_NODES;
    const float ag = sqrtf(wgam_p[0] * n);
    const float ad = sqrtf(wdel_p[0] * n);
    const float zv = zd[(size_t)i * D_LAT + k];

    float val;
    if (!isB) {
        if (s == 0)      { float g = gd_raw[(size_t)i * D_LAT + k] + SMALL_EPS; val = -ag / g; }
        else if (s == 1) { float g = gd_raw[(size_t)i * D_LAT + k] + SMALL_EPS; val = 2.0f * ag * zv / g; }
        else if (s == 2) { val = -ad * zv * zv; }
        else             { val = 2.0f * ad * zv; }
        Apk[idx] = (_Float16)val;
    } else {
        if (s == 0)      { val = ag * zv * zv; }
        else if (s == 1) { val = ag * zv; }
        else if (s == 2) { float d = dd_raw[(size_t)i * D_LAT + k] + SMALL_EPS; val = ad / d; }
        else             { float d = dd_raw[(size_t)i * D_LAT + k] + SMALL_EPS; val = ad * zv / d; }
        Bpk[idx] = (_Float16)val;
    }
}

// ---------------------------------------------------------------------------
// Kernel 5: 8192x8192 GEMM (K=512) with v_wmma_f32_16x16x32_f16 and fused
// r_i + c_j + sigmoid epilogue.
// 256 threads = 8 wave32; block tile 128x128; waves arranged 4x2, each wave
// owns a 32x64 sub-block = 2 A panels x 4 B panels = 8 accumulator tiles.
// Per K-block: 6 v16h loads feed 8 WMMAs (~21 FLOP per byte loaded per wave).  The K
// loop is fully unrolled with operands loaded straight into fresh registers
// each block — no register rotation, so the scheduler hoists next-block
// load clauses across the current WMMA chain without mov/NOP hazards.
// ---------------------------------------------------------------------------
__global__ void __launch_bounds__(256)
lsm_gemm_sigmoid(const _Float16* __restrict__ Apk,
                 const _Float16* __restrict__ Bpk,
                 const float* __restrict__ rvec,
                 const float* __restrict__ cvec,
                 float* __restrict__ out) {
    const int wave = threadIdx.x >> 5;
    const int lane = threadIdx.x & 31;
    const int wr = wave >> 1;               // 0..3 : which 32-row strip
    const int wc = wave & 1;                // 0..1 : which 64-col strip

    const int rowPanel0 = blockIdx.y * 8 + wr * 2;   // 2 consecutive 16-row panels
    const int colPanel0 = blockIdx.x * 8 + wc * 4;   // 4 consecutive 16-col panels

    const _Float16* Ab0 = Apk + (size_t)rowPanel0 * (16 * KDIM) + lane * 16;
    const _Float16* Bb0 = Bpk + (size_t)colPanel0 * (16 * KDIM) + lane * 16;
    const size_t panelStride = 16 * KDIM;   // halves per 16-node panel

    v8f acc[2][4];
    const v8f vzero = {0.f, 0.f, 0.f, 0.f, 0.f, 0.f, 0.f, 0.f};
#pragma unroll
    for (int mi = 0; mi < 2; ++mi)
#pragma unroll
        for (int ni = 0; ni < 4; ++ni) acc[mi][ni] = vzero;

#pragma unroll
    for (int kb = 0; kb < KDIM / 32; ++kb) {
        const size_t koff = (size_t)kb * 512;          // halves per K-block

        v16h a0 = *(const v16h*)(Ab0 + koff);
        v16h a1 = *(const v16h*)(Ab0 + panelStride + koff);
        v16h b0 = *(const v16h*)(Bb0 + koff);
        v16h b1 = *(const v16h*)(Bb0 + 1 * panelStride + koff);
        v16h b2 = *(const v16h*)(Bb0 + 2 * panelStride + koff);
        v16h b3 = *(const v16h*)(Bb0 + 3 * panelStride + koff);

        acc[0][0] = __builtin_amdgcn_wmma_f32_16x16x32_f16(
            false, a0, false, b0, (short)0, acc[0][0], false, false);
        acc[0][1] = __builtin_amdgcn_wmma_f32_16x16x32_f16(
            false, a0, false, b1, (short)0, acc[0][1], false, false);
        acc[0][2] = __builtin_amdgcn_wmma_f32_16x16x32_f16(
            false, a0, false, b2, (short)0, acc[0][2], false, false);
        acc[0][3] = __builtin_amdgcn_wmma_f32_16x16x32_f16(
            false, a0, false, b3, (short)0, acc[0][3], false, false);
        acc[1][0] = __builtin_amdgcn_wmma_f32_16x16x32_f16(
            false, a1, false, b0, (short)0, acc[1][0], false, false);
        acc[1][1] = __builtin_amdgcn_wmma_f32_16x16x32_f16(
            false, a1, false, b1, (short)0, acc[1][1], false, false);
        acc[1][2] = __builtin_amdgcn_wmma_f32_16x16x32_f16(
            false, a1, false, b2, (short)0, acc[1][2], false, false);
        acc[1][3] = __builtin_amdgcn_wmma_f32_16x16x32_f16(
            false, a1, false, b3, (short)0, acc[1][3], false, false);
    }

    // Epilogue: C/D layout -> lane holds rows (lane>>4)*8 + v, col lane&15.
#pragma unroll
    for (int mi = 0; mi < 2; ++mi) {
        const int rbase = (rowPanel0 + mi) * 16 + (lane >> 4) * 8;
#pragma unroll
        for (int ni = 0; ni < 4; ++ni) {
            const int col = (colPanel0 + ni) * 16 + (lane & 15);
            const float cj = cvec[col];
#pragma unroll
            for (int v = 0; v < 8; ++v) {
                const int row = rbase + v;
                const float x = acc[mi][ni][v] + rvec[row] + cj;
                out[(size_t)row * N_NODES + col] = 1.0f / (1.0f + __expf(-x));
            }
        }
    }
}

// ---------------------------------------------------------------------------
// Launch
// ---------------------------------------------------------------------------
extern "C" void kernel_launch(void* const* d_in, const int* in_sizes, int n_in,
                              void* d_out, int out_size, void* d_ws, size_t ws_size,
                              hipStream_t stream) {
    const float* z     = (const float*)d_in[0];
    const float* gamma = (const float*)d_in[1];
    const float* delta = (const float*)d_in[2];
    const float* Wz    = (const float*)d_in[3];
    const float* Wg    = (const float*)d_in[4];
    const float* Wd    = (const float*)d_in[5];
    const float* bias  = (const float*)d_in[6];
    const float* wgam  = (const float*)d_in[7];
    const float* wdel  = (const float*)d_in[8];

    float* out = (float*)d_out;
    const size_t NN = (size_t)N_NODES * N_NODES;
    float* zd = out + NN;                               // [N, 128] output #2
    float* gd = zd + (size_t)N_NODES * D_LAT;           // [N, 128] output #3
    float* dd = gd + (size_t)N_NODES * D_LAT;           // [N, 128] output #4

    // Workspace layout (all offsets 256B aligned):
    //   T    : 3*512*128 f32  =   786432 B
    //   rvec : 8192 f32       =    32768 B
    //   cvec : 8192 f32       =    32768 B
    //   Apk  : 8192*512 f16   =  8388608 B
    //   Bpk  : 8192*512 f16   =  8388608 B   (total ~16.8 MB)
    char* ws = (char*)d_ws;
    float*    T    = (float*)(ws);
    float*    rvec = (float*)(ws + 786432);
    float*    cvec = (float*)(ws + 786432 + 32768);
    _Float16* Apk  = (_Float16*)(ws + 786432 + 65536);
    _Float16* Bpk  = (_Float16*)(ws + 786432 + 65536 + 8388608);

    lsm_softmax_cols<<<384, 256, 0, stream>>>(Wz, Wg, Wd, T);
    lsm_proj<<<dim3(N_NODES / 16, 3), 128, 0, stream>>>(z, gamma, delta, T, zd, gd, dd);
    lsm_rowcol<<<N_NODES, 128, 0, stream>>>(zd, gd, dd, bias, wgam, wdel, rvec, cvec);

    const int packThreads = 2 * N_NODES * KDIM;         // A and B
    lsm_pack<<<packThreads / 256, 256, 0, stream>>>(zd, gd, dd, wgam, wdel, Apk, Bpk);

    lsm_gemm_sigmoid<<<dim3(N_NODES / 128, N_NODES / 128), 256, 0, stream>>>(
        Apk, Bpk, rvec, cvec, out);
}
